// TinyGpt_77532749627668
// MI455X (gfx1250) — matrix operand order repaired and checked by reading, not compiled, not run on hardware
//
#include <hip/hip_runtime.h>

// ---------------------------------------------------------------------------
// TinyGPT forward for gfx1250 (MI455X): bf16 WMMA GEMMs (pre-transposed
// weights, double-buffered LDS, 64x64 wave tiles, async global->LDS staging)
// + flash attention.
// ---------------------------------------------------------------------------

typedef __attribute__((ext_vector_type(16))) __bf16 v16bf;
typedef __attribute__((ext_vector_type(8)))  float  v8f;
typedef int v4i_vs __attribute__((vector_size(16)));   // int4 as plain vector

union FragB  { v16bf v; uint4 u[2]; };
union Pack16 { uint4 u[2]; __bf16 e[16]; };
union Pack8  { uint4 u;    __bf16 e[8]; };

__device__ __forceinline__ v8f wmma_bf16(const FragB& a, const FragB& b, v8f c) {
  return __builtin_amdgcn_wmma_f32_16x16x32_bf16(false, a.v, false, b.v,
                                                 (short)0, c, false, false);
}

// ---------------- async global -> LDS (CDNA5 path, guarded) ----------------
#if defined(__has_builtin)
#  if __has_builtin(__builtin_amdgcn_global_load_async_to_lds_b128)
#    define HAVE_ASYNC_LDS 1
#  endif
#endif
#ifndef HAVE_ASYNC_LDS
#  define HAVE_ASYNC_LDS 0
#endif

// copy 16 bytes (8 bf16) from global to LDS
__device__ __forceinline__ void lds_copy16(const __bf16* g, __bf16* l) {
#if HAVE_ASYNC_LDS
  __builtin_amdgcn_global_load_async_to_lds_b128(
      (__attribute__((address_space(1))) v4i_vs*)
          (__attribute__((address_space(1))) const void*)g,
      (__attribute__((address_space(3))) v4i_vs*)
          (__attribute__((address_space(3))) void*)l,
      0, 0);
#else
  *(uint4*)l = *(const uint4*)g;
#endif
}

__device__ __forceinline__ void async_wait() {
#if HAVE_ASYNC_LDS
#  if __has_builtin(__builtin_amdgcn_s_wait_asynccnt)
  __builtin_amdgcn_s_wait_asynccnt(0);
#  else
  asm volatile("s_wait_asynccnt 0x0" ::: "memory");
#  endif
#endif
}

// ---------------- fp32 [K,N] -> bf16 [N,K] (tiled transpose) ---------------
// grid: (N/64, K/64, num_matrices); per-matrix stride = K*N.
__global__ __launch_bounds__(256)
void transpose_to_bf16_kernel(const float* __restrict__ in, __bf16* __restrict__ out,
                              int K, int N) {
  __shared__ __bf16 tile[64][72];
  const float* src0 = in  + (size_t)blockIdx.z * K * N;
  __bf16*      dst0 = out + (size_t)blockIdx.z * K * N;
  int k0 = blockIdx.y * 64, n0 = blockIdx.x * 64;
  int tid = threadIdx.x;
  int r = tid >> 2;            // 0..63
  int c = (tid & 3) * 16;      // 0,16,32,48
  const float* src = src0 + (size_t)(k0 + r) * N + n0 + c;
  #pragma unroll
  for (int i = 0; i < 16; i += 4) {
    float4 v = *(const float4*)(src + i);
    tile[r][c + i + 0] = (__bf16)v.x;
    tile[r][c + i + 1] = (__bf16)v.y;
    tile[r][c + i + 2] = (__bf16)v.z;
    tile[r][c + i + 3] = (__bf16)v.w;
  }
  __syncthreads();
  Pack16 p;
  #pragma unroll
  for (int i = 0; i < 16; ++i) p.e[i] = tile[c + i][r];
  __bf16* dst = dst0 + (size_t)(n0 + r) * K + k0 + c;
  *(uint4*)(dst)     = p.u[0];
  *(uint4*)(dst + 8) = p.u[1];
}

// ------------------------------- embedding ---------------------------------
__global__ __launch_bounds__(256)
void embed_kernel(const int* __restrict__ idx, const float* __restrict__ tok,
                  const float* __restrict__ pos, float* __restrict__ x) {
  constexpr int E = 1024, T = 2048;
  int row = blockIdx.x;                 // 0 .. B*T-1
  int t   = row % T;
  int id  = idx[row];
  int c   = threadIdx.x * 4;
  float4 a = *(const float4*)(tok + (size_t)id * E + c);
  float4 p = *(const float4*)(pos + (size_t)t  * E + c);
  a.x += p.x; a.y += p.y; a.z += p.z; a.w += p.w;
  *(float4*)(x + (size_t)row * E + c) = a;
}

// ------------------------------- layernorm ---------------------------------
__global__ __launch_bounds__(256)
void ln_kernel(const float* __restrict__ x, const float* __restrict__ g,
               const float* __restrict__ bta, __bf16* __restrict__ out) {
  constexpr int E = 1024;
  int row = blockIdx.x;
  int tid = threadIdx.x;
  const float* xr = x + (size_t)row * E;
  float4 v = *(const float4*)(xr + tid * 4);
  float s  = v.x + v.y + v.z + v.w;
  float ss = v.x * v.x + v.y * v.y + v.z * v.z + v.w * v.w;
  #pragma unroll
  for (int m = 1; m < 32; m <<= 1) { s += __shfl_xor(s, m); ss += __shfl_xor(ss, m); }
  __shared__ float sums[8], sqs[8];
  if ((tid & 31) == 0) { sums[tid >> 5] = s; sqs[tid >> 5] = ss; }
  __syncthreads();
  s = 0.f; ss = 0.f;
  #pragma unroll
  for (int i = 0; i < 8; ++i) { s += sums[i]; ss += sqs[i]; }
  float mu  = s * (1.f / E);
  float var = ss * (1.f / E) - mu * mu;
  float rs  = rsqrtf(var + 1e-5f);
  int c = tid * 4;
  float vv[4] = {v.x, v.y, v.z, v.w};
  #pragma unroll
  for (int i = 0; i < 4; ++i)
    out[(size_t)row * E + c + i] = (__bf16)((vv[i] - mu) * rs * g[c + i] + bta[c + i]);
}

// --------------------------------- GEMM ------------------------------------
// C[M,N] = A[M,K](bf16,row-major) @ W, with W given PRE-TRANSPOSED as
// Wt[N,K](bf16). Block tile 256x128 (8 waves, each 64x64 = 4x4 wmma accs),
// KT=32, double-buffered LDS, async global->LDS staging.
template <bool BIAS, bool RES, bool GELU, bool OUT_BF16>
__global__ __launch_bounds__(256)
void gemm_bf16_kernel(const __bf16* __restrict__ A, const __bf16* __restrict__ Wt,
                      const float* __restrict__ bias, const float* __restrict__ res,
                      void* __restrict__ Cout, int M, int N, int K) {
  __shared__ __align__(16) __bf16 As[2][256][40];   // [buf][m][k], padded
  __shared__ __align__(16) __bf16 Bs[2][128][40];   // [buf][n][k], padded
  const int tid  = threadIdx.x;
  const int lane = tid & 31, w = tid >> 5;
  const int hh   = lane >> 4, ln = lane & 15;
  const int m0   = blockIdx.y * 256;
  const int n0   = blockIdx.x * 128;
  const int wm   = (w >> 1) * 64;   // 0,64,128,192
  const int wn   = (w & 1) * 64;    // 0,64

  v8f acc[4][4] = {};

  const int sc = (tid & 3) * 8;     // staging column (8 elems = 16B)
  const int sr = tid >> 2;          // staging row within 64-row slab

  auto stage = [&](int k0, int buf) {
    #pragma unroll
    for (int i = 0; i < 4; ++i) {   // A: 256 rows x 32 k
      int r = i * 64 + sr;
      lds_copy16(A + (size_t)(m0 + r) * K + k0 + sc, &As[buf][r][sc]);
    }
    #pragma unroll
    for (int i = 0; i < 2; ++i) {   // B: 128 rows x 32 k (from Wt[N,K])
      int r = i * 64 + sr;
      lds_copy16(Wt + (size_t)(n0 + r) * K + k0 + sc, &Bs[buf][r][sc]);
    }
  };

  const int nk = K >> 5;
  stage(0, 0);
  for (int ch = 0; ch < nk; ++ch) {
    async_wait();                       // previous chunk's async LDS writes done
    __syncthreads();                    // visible to all waves
    if (ch + 1 < nk) stage((ch + 1) << 5, (ch + 1) & 1);
    const int buf = ch & 1;

    FragB a[4], b[4];
    #pragma unroll
    for (int i = 0; i < 4; ++i) {
      int r = wm + i * 16 + ln;
      a[i].u[0] = *(const uint4*)&As[buf][r][8 * hh];
      a[i].u[1] = *(const uint4*)&As[buf][r][16 + 8 * hh];
    }
    #pragma unroll
    for (int j = 0; j < 4; ++j) {
      int c = wn + j * 16 + ln;
      b[j].u[0] = *(const uint4*)&Bs[buf][c][8 * hh];
      b[j].u[1] = *(const uint4*)&Bs[buf][c][16 + 8 * hh];
    }
    #pragma unroll
    for (int i = 0; i < 4; ++i)
      #pragma unroll
      for (int j = 0; j < 4; ++j)
        acc[i][j] = wmma_bf16(a[i], b[j], acc[i][j]);
  }

  // epilogue: C elem e -> row = base + e + 8*hh, col = base + ln
  #pragma unroll
  for (int i = 0; i < 4; ++i) {
    #pragma unroll
    for (int j = 0; j < 4; ++j) {
      int col  = n0 + wn + j * 16 + ln;
      float bv = BIAS ? bias[col] : 0.f;
      #pragma unroll
      for (int e = 0; e < 8; ++e) {
        int row = m0 + wm + i * 16 + e + 8 * hh;
        float v = acc[i][j][e] + bv;
        if (RES)  v += res[(size_t)row * N + col];
        if (GELU) v = 0.5f * v * (1.f + erff(v * 0.70710678118654752f));
        if (OUT_BF16) ((__bf16*)Cout)[(size_t)row * N + col] = (__bf16)v;
        else          ((float*)Cout)[(size_t)row * N + col]  = v;
      }
    }
  }
}

// ---------------------------- flash attention ------------------------------
// One block = (b, h, 128 query rows), 8 waves x 16 rows each. Causal, D=64.
__global__ __launch_bounds__(256)
void flash_attn_kernel(const __bf16* __restrict__ Q, const __bf16* __restrict__ Kg,
                       const __bf16* __restrict__ Vg, __bf16* __restrict__ O) {
  constexpr int T = 2048, E = 1024, D = 64, NB = T / 128;
  __shared__ __align__(16) __bf16 Ks[32][72];      // [s_local][d]
  __shared__ __align__(16) __bf16 Vt[64][40];      // [d][s_local]
  __shared__ __align__(16) __bf16 Pl[8][16][40];   // per-wave P (16 rows x 32 s)

  int blk = blockIdx.x;
  int mb  = blk % NB;
  int h   = (blk / NB) % 16;
  int b   = blk / (NB * 16);
  int rowBase = mb * 128;
  int tid = threadIdx.x, lane = tid & 31, w = tid >> 5;
  int hh = lane >> 4, ln = lane & 15;
  int m0 = rowBase + w * 16;

  // Q fragments (A-layout), d split into [0,32) and [32,64)
  size_t qoff = (size_t)(b * T + m0 + ln) * E + h * D;
  FragB qa0, qa1;
  qa0.u[0] = *(const uint4*)(Q + qoff + 8 * hh);
  qa0.u[1] = *(const uint4*)(Q + qoff + 16 + 8 * hh);
  qa1.u[0] = *(const uint4*)(Q + qoff + 32 + 8 * hh);
  qa1.u[1] = *(const uint4*)(Q + qoff + 32 + 16 + 8 * hh);

  float mrow[8], lrow[8];
  v8f oacc[4] = {};
  #pragma unroll
  for (int j = 0; j < 8; ++j) { mrow[j] = -1e30f; lrow[j] = 0.f; }

  int nch = rowBase / 32 + 4;   // chunks of 32 keys, up to the block diagonal
  for (int ch = 0; ch < nch; ++ch) {
    int s0 = ch * 32;
    { // cooperative staging: K async-copied, V transposed through VGPRs
      int sl = tid >> 3, dbase = (tid & 7) * 8;
      size_t goff = (size_t)(b * T + s0 + sl) * E + h * D + dbase;
      lds_copy16(Kg + goff, &Ks[sl][dbase]);
      Pack8 p; p.u = *(const uint4*)(Vg + goff);
      #pragma unroll
      for (int i = 0; i < 8; ++i) Vt[dbase + i][sl] = p.e[i];
    }
    async_wait();
    __syncthreads();

    // S = Q K^T for two 16-col tiles (D=64 -> 2 chained wmmas each)
    v8f sA = {}, sB = {};
    FragB kb;
    kb.u[0] = *(const uint4*)&Ks[ln][8 * hh];
    kb.u[1] = *(const uint4*)&Ks[ln][16 + 8 * hh];
    sA = wmma_bf16(qa0, kb, sA);
    kb.u[0] = *(const uint4*)&Ks[ln][32 + 8 * hh];
    kb.u[1] = *(const uint4*)&Ks[ln][32 + 16 + 8 * hh];
    sA = wmma_bf16(qa1, kb, sA);
    kb.u[0] = *(const uint4*)&Ks[16 + ln][8 * hh];
    kb.u[1] = *(const uint4*)&Ks[16 + ln][16 + 8 * hh];
    sB = wmma_bf16(qa0, kb, sB);
    kb.u[0] = *(const uint4*)&Ks[16 + ln][32 + 8 * hh];
    kb.u[1] = *(const uint4*)&Ks[16 + ln][32 + 16 + 8 * hh];
    sB = wmma_bf16(qa1, kb, sB);

    // online softmax: row of element j is (m0 + j + 8*hh); cols live across ln
    float p0[8], p1[8], alpha[8];
    #pragma unroll
    for (int j = 0; j < 8; ++j) {
      int row = m0 + j + 8 * hh;
      float v0 = sA[j] * 0.125f; if (s0 + ln      > row) v0 = -1e30f;
      float v1 = sB[j] * 0.125f; if (s0 + 16 + ln > row) v1 = -1e30f;
      float r = fmaxf(v0, v1);
      r = fmaxf(r, __shfl_xor(r, 1)); r = fmaxf(r, __shfl_xor(r, 2));
      r = fmaxf(r, __shfl_xor(r, 4)); r = fmaxf(r, __shfl_xor(r, 8));
      float mn = fmaxf(mrow[j], r);
      alpha[j] = __expf(mrow[j] - mn);
      mrow[j]  = mn;
      p0[j] = __expf(v0 - mn);
      p1[j] = __expf(v1 - mn);
      float ps = p0[j] + p1[j];
      ps += __shfl_xor(ps, 1); ps += __shfl_xor(ps, 2);
      ps += __shfl_xor(ps, 4); ps += __shfl_xor(ps, 8);
      lrow[j] = lrow[j] * alpha[j] + ps;
    }
    #pragma unroll
    for (int dt = 0; dt < 4; ++dt)
      #pragma unroll
      for (int j = 0; j < 8; ++j) oacc[dt][j] *= alpha[j];

    // P (C-layout) -> LDS -> A-layout fragment
    #pragma unroll
    for (int j = 0; j < 8; ++j) {
      Pl[w][j + 8 * hh][ln]      = (__bf16)p0[j];
      Pl[w][j + 8 * hh][16 + ln] = (__bf16)p1[j];
    }
    FragB pa;
    pa.u[0] = *(const uint4*)&Pl[w][ln][8 * hh];
    pa.u[1] = *(const uint4*)&Pl[w][ln][16 + 8 * hh];

    #pragma unroll
    for (int dt = 0; dt < 4; ++dt) {
      FragB vb;
      vb.u[0] = *(const uint4*)&Vt[dt * 16 + ln][8 * hh];
      vb.u[1] = *(const uint4*)&Vt[dt * 16 + ln][16 + 8 * hh];
      oacc[dt] = wmma_bf16(pa, vb, oacc[dt]);
    }
    __syncthreads();
  }

  float inv[8];
  #pragma unroll
  for (int j = 0; j < 8; ++j) inv[j] = 1.f / lrow[j];
  #pragma unroll
  for (int dt = 0; dt < 4; ++dt)
    #pragma unroll
    for (int j = 0; j < 8; ++j) {
      size_t row = (size_t)(b * T + m0 + j + 8 * hh);
      O[row * E + h * D + dt * 16 + ln] = (__bf16)(oacc[dt][j] * inv[j]);
    }
}

// ------------------------------- launcher ----------------------------------
extern "C" void kernel_launch(void* const* d_in, const int* in_sizes, int n_in,
                              void* d_out, int out_size, void* d_ws, size_t ws_size,
                              hipStream_t stream) {
  (void)in_sizes; (void)n_in; (void)out_size; (void)ws_size;
  constexpr int V = 32000, E = 1024, L = 8, T = 2048, B = 2;
  constexpr int M = B * T, F = 4 * E;

  const int*   indices = (const int*)  d_in[0];
  const float* tok     = (const float*)d_in[1];
  const float* pos     = (const float*)d_in[2];
  const float* Wq      = (const float*)d_in[3];
  const float* Wk      = (const float*)d_in[4];
  const float* Wv      = (const float*)d_in[5];
  const float* Wp      = (const float*)d_in[6];
  const float* bp      = (const float*)d_in[7];
  const float* ln1_g   = (const float*)d_in[8];
  const float* ln1_b   = (const float*)d_in[9];
  const float* ln2_g   = (const float*)d_in[10];
  const float* ln2_b   = (const float*)d_in[11];
  const float* W1      = (const float*)d_in[12];
  const float* b1      = (const float*)d_in[13];
  const float* W2      = (const float*)d_in[14];
  const float* b2      = (const float*)d_in[15];
  const float* lnf_g   = (const float*)d_in[16];
  const float* lnf_b   = (const float*)d_in[17];
  const float* Wout    = (const float*)d_in[18];

  char* base = (char*)d_ws;
  size_t off = 0;
  auto carve = [&](size_t bytes) -> void* {
    void* p = base + off;
    off += (bytes + 255) & ~(size_t)255;
    return p;
  };

  float*  x    = (float*) carve((size_t)M * E * 4);
  __bf16* hb   = (__bf16*)carve((size_t)M * E * 2);
  __bf16* qb   = (__bf16*)carve((size_t)M * E * 2);
  __bf16* kbuf = (__bf16*)carve((size_t)M * E * 2);
  __bf16* vbuf = (__bf16*)carve((size_t)M * E * 2);
  __bf16* ob   = (__bf16*)carve((size_t)M * E * 2);
  __bf16* midb = (__bf16*)carve((size_t)M * F * 2);
  // transposed bf16 weights: Wt[l] is [N,K]
  __bf16* WqT  = (__bf16*)carve((size_t)L * E * E * 2);
  __bf16* WkT  = (__bf16*)carve((size_t)L * E * E * 2);
  __bf16* WvT  = (__bf16*)carve((size_t)L * E * E * 2);
  __bf16* WpT  = (__bf16*)carve((size_t)L * E * E * 2);
  __bf16* W1T  = (__bf16*)carve((size_t)L * E * F * 2);
  __bf16* W2T  = (__bf16*)carve((size_t)L * F * E * 2);
  __bf16* WoT  = (__bf16*)carve((size_t)E * V * 2);

  // Pre-transpose + convert all weights (once per launch; bandwidth-trivial).
  transpose_to_bf16_kernel<<<dim3(E / 64, E / 64, L), 256, 0, stream>>>(Wq, WqT, E, E);
  transpose_to_bf16_kernel<<<dim3(E / 64, E / 64, L), 256, 0, stream>>>(Wk, WkT, E, E);
  transpose_to_bf16_kernel<<<dim3(E / 64, E / 64, L), 256, 0, stream>>>(Wv, WvT, E, E);
  transpose_to_bf16_kernel<<<dim3(E / 64, E / 64, L), 256, 0, stream>>>(Wp, WpT, E, E);
  transpose_to_bf16_kernel<<<dim3(F / 64, E / 64, L), 256, 0, stream>>>(W1, W1T, E, F);
  transpose_to_bf16_kernel<<<dim3(E / 64, F / 64, L), 256, 0, stream>>>(W2, W2T, F, E);
  transpose_to_bf16_kernel<<<dim3(V / 64, E / 64, 1), 256, 0, stream>>>(Wout, WoT, E, V);

  embed_kernel<<<M, 256, 0, stream>>>(indices, tok, pos, x);

  dim3 blk(256);
  dim3 gEE(E / 128, M / 256);   // N=1024
  dim3 gEF(F / 128, M / 256);   // N=4096
  dim3 gEV(V / 128, M / 256);   // N=32000

  for (int l = 0; l < L; ++l) {
    ln_kernel<<<M, 256, 0, stream>>>(x, ln1_g + (size_t)l * E, ln1_b + (size_t)l * E, hb);
    gemm_bf16_kernel<false, false, false, true><<<gEE, blk, 0, stream>>>(
        hb, WqT + (size_t)l * E * E, nullptr, nullptr, qb, M, E, E);
    gemm_bf16_kernel<false, false, false, true><<<gEE, blk, 0, stream>>>(
        hb, WkT + (size_t)l * E * E, nullptr, nullptr, kbuf, M, E, E);
    gemm_bf16_kernel<false, false, false, true><<<gEE, blk, 0, stream>>>(
        hb, WvT + (size_t)l * E * E, nullptr, nullptr, vbuf, M, E, E);

    flash_attn_kernel<<<B * 16 * (T / 128), 256, 0, stream>>>(qb, kbuf, vbuf, ob);

    gemm_bf16_kernel<true, true, false, false><<<gEE, blk, 0, stream>>>(
        ob, WpT + (size_t)l * E * E, bp + (size_t)l * E, x, x, M, E, E);

    ln_kernel<<<M, 256, 0, stream>>>(x, ln2_g + (size_t)l * E, ln2_b + (size_t)l * E, hb);
    gemm_bf16_kernel<true, false, true, true><<<gEF, blk, 0, stream>>>(
        hb, W1T + (size_t)l * E * F, b1 + (size_t)l * F, nullptr, midb, M, F, E);
    gemm_bf16_kernel<true, true, false, false><<<gEE, blk, 0, stream>>>(
        midb, W2T + (size_t)l * F * E, b2 + (size_t)l * E, x, x, M, E, F);
  }

  ln_kernel<<<M, 256, 0, stream>>>(x, lnf_g, lnf_b, hb);
  gemm_bf16_kernel<false, false, false, false><<<gEV, blk, 0, stream>>>(
      hb, WoT, nullptr, nullptr, d_out, M, V, E);
}
